// MemoryMultiHeadAttention_91216515433118
// MI455X (gfx1250) — matrix operand order repaired and checked by reading, not compiled, not run on hardware
//
#include <hip/hip_runtime.h>
#include <stdint.h>

// ---------------------------------------------------------------------------
// MemoryMultiHeadAttention for MI455X (gfx1250, wave32, WMMA bf16 16x16x32)
//
// Data movement paths exercised:
//   - TDM (tensor_load_to_lds + s_wait_tensorcnt): attention K tiles
//   - async global->LDS (global_load_async_to_lds_b128 + s_wait_asynccnt): GEMM tiles
//   - manual ds staging: fp32->bf16 converting loads, V transpose
// Both GEMM and attention are double-buffered: next tile's copies are issued
// before computing the current tile.
// ---------------------------------------------------------------------------

typedef __attribute__((ext_vector_type(16))) __bf16 v16bf;
typedef __attribute__((ext_vector_type(8)))  __bf16 bf16x8;
typedef __attribute__((ext_vector_type(8)))  float  v8f;
typedef __attribute__((ext_vector_type(4)))  unsigned int u32x4;
typedef __attribute__((ext_vector_type(4)))  int i32x4;
typedef __attribute__((ext_vector_type(8)))  int i32x8;

#define WMMA_BF16(A, B, C) \
  __builtin_amdgcn_wmma_f32_16x16x32_bf16(false, (A), false, (B), (short)0, (C), false, false)

static __device__ __forceinline__ v16bf frag2(const __bf16* p0, const __bf16* p1) {
  bf16x8 lo = *(const bf16x8*)p0;
  bf16x8 hi = *(const bf16x8*)p1;
  return __builtin_shufflevector(lo, hi, 0,1,2,3,4,5,6,7,8,9,10,11,12,13,14,15);
}

// Async 16-byte global -> LDS copy (per active lane). ASYNCcnt-tracked.
static __device__ __forceinline__ void async_copy_b128(uint32_t lds_off, const void* gptr) {
  uint64_t ga = (uint64_t)(uintptr_t)gptr;
  asm volatile("global_load_async_to_lds_b128 %0, %1, off" :: "v"(lds_off), "v"(ga) : "memory");
}
static __device__ __forceinline__ void wait_asynccnt0() {
  asm volatile("s_wait_asynccnt 0x0" ::: "memory");
}
static __device__ __forceinline__ uint32_t lds_offset(const void* p) {
  return (uint32_t)(uintptr_t)p;  // low 32 bits of generic ptr = LDS byte offset
}

// ---- Tensor Data Mover: 2D bf16 tile (tile_d0 x tile_d1) global -> LDS ----
// D# group1 is tile-geometry only (uniform); group0 carries lds/global addr.
static __device__ __forceinline__ i32x8 tdm_group1_2d(int tensor_d0, int tensor_d1,
                                                      int stride0, int tile_d0, int tile_d1) {
  i32x8 g1;
  g1[0] = 1 << 16;                                   // data_size = 2 bytes
  g1[1] = (tensor_d0 & 0xFFFF) << 16;                // tensor_dim0[15:0] @ bit48
  g1[2] = ((tensor_d0 >> 16) & 0xFFFF) | ((tensor_d1 & 0xFFFF) << 16);
  g1[3] = ((tensor_d1 >> 16) & 0xFFFF) | (tile_d0 << 16);  // tile_dim0 @ bit112
  g1[4] = tile_d1;                                   // tile_dim1; tile_dim2 = 0
  g1[5] = stride0;                                   // tensor_dim0_stride[31:0]
  g1[6] = 0;                                         // stride0 hi16 = 0; dim1_stride = 0
  g1[7] = 0;
  return g1;
}

static __device__ __forceinline__ void tdm_load_2d(uint32_t lds_addr, const void* gptr, i32x8 g1) {
  uint64_t ga = (uint64_t)(uintptr_t)gptr;
  u32x4 g0;
  g0[0] = 1u;                                        // count=1, user descriptor
  g0[1] = lds_addr;                                  // lds_addr @ bits[63:32]
  g0[2] = (uint32_t)ga;                              // global_addr[31:0]
  g0[3] = ((uint32_t)(ga >> 32) & 0x01FFFFFFu) | 0x80000000u;  // addr[56:32] | type=2
  i32x4 z4 = {0, 0, 0, 0};
#if __clang_major__ >= 23
  i32x8 z8 = {0, 0, 0, 0, 0, 0, 0, 0};
  __builtin_amdgcn_tensor_load_to_lds(g0, g1, z4, z4, z8, 0);
#else
  __builtin_amdgcn_tensor_load_to_lds(g0, g1, z4, z4, 0);
#endif
}
static __device__ __forceinline__ void wait_tensorcnt0() {
  __builtin_amdgcn_s_wait_tensorcnt(0);
}

// ------------------------------- constants ---------------------------------
#define C_B    2
#define C_S    2048
#define C_H    16
#define C_DK   64
#define C_DV   64
#define C_MEM  128
#define C_T    2176        // S + MEM
#define C_DM   1024        // D_MODEL = H*DK = H*DV = OUT_DIM
#define C_ROWS 4096        // B * S

// --------------------------- convert kernels -------------------------------
__global__ void cvt_bf16_kernel(const float* __restrict__ in, __bf16* __restrict__ out, int n) {
  int i = blockIdx.x * 256 + threadIdx.x;
  if (i < n) out[i] = (__bf16)in[i];
}

// in: fp32 [K,N] row-major  ->  out: bf16 [N,K] row-major (transposed)
__global__ void cvtT_bf16_kernel(const float* __restrict__ in, __bf16* __restrict__ out, int K, int N) {
  int i = blockIdx.x * 256 + threadIdx.x;
  if (i < K * N) {
    int n = i / K, k = i - n * K;
    out[i] = (__bf16)in[(size_t)k * N + n];
  }
}

// ------------------------------ GEMM kernel --------------------------------
// C[M,N] = A[M,K] @ Bt[N,K]^T + bias[N]
// block tile 128(M) x 64(N), BK = 32, 256 threads = 8 waves (4x2), wave 32x32
// Double-buffered LDS; bf16 tiles staged with async global->LDS copies.
template <bool A_F32, bool OUT_F32>
__global__ __launch_bounds__(256) void gemm_bf16_kernel(const void* __restrict__ Avoid,
                                                        const __bf16* __restrict__ Bt,
                                                        const float* __restrict__ bias,
                                                        void* __restrict__ Cout,
                                                        int M, int N, int K) {
  __shared__ __bf16 As[2][128 * 32];
  __shared__ __bf16 Bs[2][64 * 32];

  const int m0 = blockIdx.y * 128;
  const int n0 = blockIdx.x * 64;
  const int tid  = threadIdx.x;
  const int lane = tid & 31;
  const int w    = tid >> 5;
  const int wm   = (w & 3) * 32;   // wave M offset in block tile
  const int wn   = (w >> 2) * 32;  // wave N offset in block tile
  const int grp  = lane >> 4;      // 0/1 half-wave
  const int ln   = lane & 15;

  auto stageA = [&](int k0, int buf) {
    if constexpr (A_F32) {
      // fp32 -> bf16 conversion in flight (manual path)
      #pragma unroll
      for (int it = 0; it < 2; ++it) {
        int i = tid + it * 256;            // 512 chunks of 8 elements
        int row = i >> 2, ch = (i & 3) * 8;
        const float* ap = (const float*)Avoid + (size_t)(m0 + row) * K + k0 + ch;
        float4 x0 = *(const float4*)ap;
        float4 x1 = *(const float4*)(ap + 4);
        bf16x8 v;
        v[0] = (__bf16)x0.x; v[1] = (__bf16)x0.y; v[2] = (__bf16)x0.z; v[3] = (__bf16)x0.w;
        v[4] = (__bf16)x1.x; v[5] = (__bf16)x1.y; v[6] = (__bf16)x1.z; v[7] = (__bf16)x1.w;
        *(bf16x8*)&As[buf][row * 32 + ch] = v;
      }
    } else {
      // raw bf16: async global -> LDS (no VGPR round trip)
      const uint32_t asBase = lds_offset(&As[buf][0]);
      #pragma unroll
      for (int it = 0; it < 2; ++it) {
        int i = tid + it * 256;
        int row = i >> 2, ch = (i & 3) * 8;
        async_copy_b128(asBase + (uint32_t)(row * 64 + ch * 2),
                        (const __bf16*)Avoid + (size_t)(m0 + row) * K + k0 + ch);
      }
    }
  };
  auto stageB = [&](int k0, int buf) {
    const uint32_t bsBase = lds_offset(&Bs[buf][0]);
    int row = tid >> 2, ch = (tid & 3) * 8;
    async_copy_b128(bsBase + (uint32_t)(row * 64 + ch * 2),
                    &Bt[(size_t)(n0 + row) * K + k0 + ch]);
  };

  v8f acc[2][2] = {};

  // prologue: stage tile 0
  stageA(0, 0);
  stageB(0, 0);
  wait_asynccnt0();
  __syncthreads();

  int cur = 0;
  for (int k0 = 0; k0 < K; k0 += 32) {
    const int nxt = cur ^ 1;
    if (k0 + 32 < K) {               // issue next tile's copies before computing
      stageA(k0 + 32, nxt);
      stageB(k0 + 32, nxt);
      __builtin_prefetch(&Bt[(size_t)(n0 + (tid >> 2)) * K + k0 + 64], 0, 0);
    }

    // ---- fragments + 4 WMMAs per wave (current buffer) ----
    v16bf af[2], bfr[2];
    #pragma unroll
    for (int i = 0; i < 2; ++i) {
      const __bf16* ap = &As[cur][(wm + i * 16 + ln) * 32 + grp * 8];
      af[i] = frag2(ap, ap + 16);
      const __bf16* bp = &Bs[cur][(wn + i * 16 + ln) * 32 + grp * 8];
      bfr[i] = frag2(bp, bp + 16);
    }
    #pragma unroll
    for (int i = 0; i < 2; ++i)
      #pragma unroll
      for (int j = 0; j < 2; ++j)
        acc[i][j] = WMMA_BF16(af[i], bfr[j], acc[i][j]);

    wait_asynccnt0();
    __syncthreads();
    cur = nxt;
  }

  // ---- epilogue: bias + store ----
  #pragma unroll
  for (int j = 0; j < 2; ++j) {
    const int n = n0 + wn + j * 16 + ln;
    const float bb = bias[n];
    #pragma unroll
    for (int i = 0; i < 2; ++i) {
      #pragma unroll
      for (int r = 0; r < 8; ++r) {
        const int m = m0 + wm + i * 16 + grp * 8 + r;
        const float v = acc[i][j][r] + bb;
        if constexpr (OUT_F32) ((float*)Cout)[(size_t)m * N + n] = v;
        else                   ((__bf16*)Cout)[(size_t)m * N + n] = (__bf16)v;
      }
    }
  }
}

// --------------------------- attention kernel ------------------------------
// grid (B*H, S/64), block 128 = 4 waves; wave owns 16 query rows.
// K tiles arrive via TDM (one DMA per block, issued by wave 0); V tiles are
// manually transposed into LDS. Both are double-buffered across the T loop.
__global__ __launch_bounds__(128) void attn_kernel(const __bf16* __restrict__ qp,
                                                   const __bf16* __restrict__ kp,
                                                   const __bf16* __restrict__ vp,
                                                   const __bf16* __restrict__ mkb,
                                                   const __bf16* __restrict__ mvb,
                                                   __bf16* __restrict__ ao) {
  __shared__ __bf16 Ks[2][64][64];       // K tile [t][dk], TDM-staged
  __shared__ __bf16 Vt[2][64][64];       // V tile transposed: Vt[dv][t]
  __shared__ __bf16 Pbuf[4][16][64];     // per-wave P tile (C-layout -> A-layout)

  const int bh = blockIdx.x;
  const int b  = bh >> 4;
  const int h  = bh & 15;
  const int q0 = blockIdx.y * 64;
  const int tid  = threadIdx.x;
  const int lane = tid & 31;
  const int w    = tid >> 5;
  const int grp  = lane >> 4;
  const int ln   = lane & 15;

  const i32x8 tdmG1 = tdm_group1_2d(/*tensor_d0=*/C_DM, /*tensor_d1=*/64,
                                    /*stride0=*/C_DM, /*tile_d0=*/64, /*tile_d1=*/64);

  auto kv_src = [&](int t0, const __bf16*& ksrc, const __bf16*& vsrc) {
    if (t0 < C_S) {
      ksrc = kp + (size_t)(b * C_S + t0) * C_DM + h * 64;
      vsrc = vp + (size_t)(b * C_S + t0) * C_DM + h * 64;
    } else {
      ksrc = mkb + (size_t)(t0 - C_S) * C_DM + h * 64;
      vsrc = mvb + (size_t)(t0 - C_S) * C_DM + h * 64;
    }
  };

  auto stageKV = [&](int t0, int buf) {
    const __bf16 *ksrc, *vsrc;
    kv_src(t0, ksrc, vsrc);
    if (w == 0) {                       // one TDM DMA moves the whole K tile
      tdm_load_2d(lds_offset(&Ks[buf][0][0]), ksrc, tdmG1);
    }
    #pragma unroll
    for (int it = 0; it < 4; ++it) {    // V: transpose on store
      int i = tid + it * 128;           // 512 chunks of 8 along dv
      int t = i >> 3, dv0 = (i & 7) * 8;
      bf16x8 vv = *(const bf16x8*)(vsrc + (size_t)t * C_DM + dv0);
      #pragma unroll
      for (int e = 0; e < 8; ++e) Vt[buf][dv0 + e][t] = vv[e];
    }
  };

  // preload Q fragments for this wave's 16 rows (A-layout, direct from global)
  const __bf16* qbase = qp + (size_t)(b * C_S + q0 + w * 16 + ln) * C_DM + h * 64;
  v16bf qa[2];
  #pragma unroll
  for (int ks = 0; ks < 2; ++ks)
    qa[ks] = frag2(qbase + ks * 32 + grp * 8, qbase + ks * 32 + 16 + grp * 8);

  float m_r[8], l_r[8];
  #pragma unroll
  for (int r = 0; r < 8; ++r) { m_r[r] = -1e30f; l_r[r] = 0.f; }
  v8f acc[4] = {};

  const float scale = 0.125f;  // 1/sqrt(DK=64)

  // prologue: stage tile 0
  stageKV(0, 0);
  if (w == 0) wait_tensorcnt0();
  __syncthreads();

  int cur = 0;
  for (int t0 = 0; t0 < C_T; t0 += 64) {
    const int nxt = cur ^ 1;
    if (t0 + 64 < C_T) stageKV(t0 + 64, nxt);  // overlap next tile's DMA

    // ---- scores: S[16,64] = Q @ K^T  (K-frags contiguous from LDS) ----
    v8f s[4] = {};
    #pragma unroll
    for (int ks = 0; ks < 2; ++ks) {
      #pragma unroll
      for (int j = 0; j < 4; ++j) {
        const __bf16* kb = &Ks[cur][j * 16 + ln][ks * 32 + grp * 8];
        v16bf bfr = frag2(kb, kb + 16);
        s[j] = WMMA_BF16(qa[ks], bfr, s[j]);
      }
    }
    #pragma unroll
    for (int j = 0; j < 4; ++j) s[j] = s[j] * scale;

    // ---- online softmax (row stats across 16-lane halves) ----
    #pragma unroll
    for (int r = 0; r < 8; ++r) {
      float sm = fmaxf(fmaxf(s[0][r], s[1][r]), fmaxf(s[2][r], s[3][r]));
      sm = fmaxf(sm, __shfl_xor(sm, 1, 32));
      sm = fmaxf(sm, __shfl_xor(sm, 2, 32));
      sm = fmaxf(sm, __shfl_xor(sm, 4, 32));
      sm = fmaxf(sm, __shfl_xor(sm, 8, 32));
      const float mnew  = fmaxf(m_r[r], sm);
      const float alpha = __expf(m_r[r] - mnew);
      m_r[r] = mnew;
      float ps = 0.f;
      #pragma unroll
      for (int j = 0; j < 4; ++j) {
        const float p = __expf(s[j][r] - mnew);
        s[j][r] = p;
        ps += p;
      }
      ps += __shfl_xor(ps, 1, 32);
      ps += __shfl_xor(ps, 2, 32);
      ps += __shfl_xor(ps, 4, 32);
      ps += __shfl_xor(ps, 8, 32);
      l_r[r] = l_r[r] * alpha + ps;
      #pragma unroll
      for (int dj = 0; dj < 4; ++dj) acc[dj][r] = acc[dj][r] * alpha;
    }

    // ---- P: C-layout regs -> LDS -> A-layout frags (wave-private region) ----
    #pragma unroll
    for (int r = 0; r < 8; ++r)
      #pragma unroll
      for (int j = 0; j < 4; ++j)
        Pbuf[w][grp * 8 + r][j * 16 + ln] = (__bf16)s[j][r];
    asm volatile("s_wait_dscnt 0" ::: "memory");  // same-wave LDS RAW

    v16bf pa[2];
    #pragma unroll
    for (int ks = 0; ks < 2; ++ks)
      pa[ks] = frag2(&Pbuf[w][ln][ks * 32 + grp * 8],
                     &Pbuf[w][ln][ks * 32 + 16 + grp * 8]);

    // ---- O += P @ V  (V-frags contiguous from transposed LDS tile) ----
    #pragma unroll
    for (int ks = 0; ks < 2; ++ks) {
      #pragma unroll
      for (int dj = 0; dj < 4; ++dj) {
        v16bf vf = frag2(&Vt[cur][dj * 16 + ln][ks * 32 + grp * 8],
                         &Vt[cur][dj * 16 + ln][ks * 32 + 16 + grp * 8]);
        acc[dj] = WMMA_BF16(pa[ks], vf, acc[dj]);
      }
    }

    if (w == 0) wait_tensorcnt0();     // next K tile landed
    __syncthreads();                   // publish Ks/Vt[nxt], retire Ks/Vt[cur]
    cur = nxt;
  }

  // ---- epilogue: normalize, store bf16 [B*S, H*DV] ----
  __bf16* obase = ao + (size_t)(b * C_S + q0 + w * 16) * C_DM + h * 64;
  #pragma unroll
  for (int r = 0; r < 8; ++r) {
    const float inv = 1.0f / l_r[r];
    #pragma unroll
    for (int dj = 0; dj < 4; ++dj)
      obase[(size_t)(grp * 8 + r) * C_DM + dj * 16 + ln] = (__bf16)(acc[dj][r] * inv);
  }
}

// ------------------------------- launcher ----------------------------------
extern "C" void kernel_launch(void* const* d_in, const int* in_sizes, int n_in,
                              void* d_out, int out_size, void* d_ws, size_t ws_size,
                              hipStream_t stream) {
  const float* queries = (const float*)d_in[0];
  const float* keys    = (const float*)d_in[1];
  const float* values  = (const float*)d_in[2];
  const float* wq_w    = (const float*)d_in[3];
  const float* wq_b    = (const float*)d_in[4];
  const float* wk_w    = (const float*)d_in[5];
  const float* wk_b    = (const float*)d_in[6];
  const float* wv_w    = (const float*)d_in[7];
  const float* wv_b    = (const float*)d_in[8];
  const float* mk      = (const float*)d_in[9];
  const float* mv      = (const float*)d_in[10];
  const float* wo_w    = (const float*)d_in[11];
  const float* wo_b    = (const float*)d_in[12];

  char* ws = (char*)d_ws;
  size_t off = 0;
  auto alloc_bf16 = [&](size_t elems) {
    __bf16* p = (__bf16*)(ws + off);
    off += elems * sizeof(__bf16);
    return p;
  };
  __bf16* wqT = alloc_bf16((size_t)C_DM * C_DM);
  __bf16* wkT = alloc_bf16((size_t)C_DM * C_DM);
  __bf16* wvT = alloc_bf16((size_t)C_DM * C_DM);
  __bf16* woT = alloc_bf16((size_t)C_DM * C_DM);
  __bf16* mkb = alloc_bf16((size_t)C_MEM * C_DM);
  __bf16* mvb = alloc_bf16((size_t)C_MEM * C_DM);
  __bf16* qpj = alloc_bf16((size_t)C_ROWS * C_DM);
  __bf16* kpj = alloc_bf16((size_t)C_ROWS * C_DM);
  __bf16* vpj = alloc_bf16((size_t)C_ROWS * C_DM);
  __bf16* aout = alloc_bf16((size_t)C_ROWS * C_DM);  // ~40.5 MB total

  // 1) weight transpose-convert + memory slot convert
  {
    const int nW = C_DM * C_DM;
    cvtT_bf16_kernel<<<(nW + 255) / 256, 256, 0, stream>>>(wq_w, wqT, C_DM, C_DM);
    cvtT_bf16_kernel<<<(nW + 255) / 256, 256, 0, stream>>>(wk_w, wkT, C_DM, C_DM);
    cvtT_bf16_kernel<<<(nW + 255) / 256, 256, 0, stream>>>(wv_w, wvT, C_DM, C_DM);
    cvtT_bf16_kernel<<<(nW + 255) / 256, 256, 0, stream>>>(wo_w, woT, C_DM, C_DM);
    const int nM = C_MEM * C_DM;
    cvt_bf16_kernel<<<(nM + 255) / 256, 256, 0, stream>>>(mk, mkb, nM);
    cvt_bf16_kernel<<<(nM + 255) / 256, 256, 0, stream>>>(mv, mvb, nM);
  }

  // 2) projections: [4096,1024] @ [1024,1024] + bias -> bf16
  dim3 ggrid(C_DM / 64, C_ROWS / 128);
  gemm_bf16_kernel<true, false><<<ggrid, 256, 0, stream>>>(queries, wqT, wq_b, qpj, C_ROWS, C_DM, C_DM);
  gemm_bf16_kernel<true, false><<<ggrid, 256, 0, stream>>>(keys,    wkT, wk_b, kpj, C_ROWS, C_DM, C_DM);
  gemm_bf16_kernel<true, false><<<ggrid, 256, 0, stream>>>(values,  wvT, wv_b, vpj, C_ROWS, C_DM, C_DM);

  // 3) flash attention over T = S + MEM (TDM-staged K tiles)
  attn_kernel<<<dim3(C_B * C_H, C_S / 64), 128, 0, stream>>>(qpj, kpj, vpj, mkb, mvb, aout);

  // 4) output projection -> fp32 d_out
  gemm_bf16_kernel<false, true><<<ggrid, 256, 0, stream>>>(aout, woT, wo_b, d_out, C_ROWS, C_DM, C_DM);
}
